// HierarchicalLFQHVQVAE_25409026523976
// MI455X (gfx1250) — compile-verified, matrix operand order checked by reading, not run on hardware
//
#include <hip/hip_runtime.h>
#include <cstdint>

#define DEVINL static __device__ __forceinline__

typedef __bf16 bf16_t;
typedef __attribute__((ext_vector_type(16))) __bf16 v16bf;
typedef __attribute__((ext_vector_type(8))) float v8f;
typedef __attribute__((ext_vector_type(4))) unsigned int v4u;
typedef __attribute__((ext_vector_type(4))) int v4i;
typedef __attribute__((ext_vector_type(8))) int v8i;

#if defined(__has_builtin)
#if __has_builtin(__builtin_amdgcn_tensor_load_to_lds) && \
    __has_builtin(__builtin_amdgcn_s_wait_tensorcnt)
#define USE_TDM 1
#endif
#endif
#ifndef USE_TDM
#define USE_TDM 0
#endif

constexpr int Bb = 64, Ss = 1024, FD = 512, ZD = 128, QD = 64, HID = 128;
constexpr int T = Bb * Ss;          // 65536 tokens
constexpr int NZ = 1024, NQ = 512;
constexpr float DECAY = 0.99f;
constexpr float EPSV = 1e-5f;
constexpr float ZTH = 3.0f, QTH = 1.0f;

// ---- output layout (floats, concatenated in reference return order) ----
constexpr size_t O_XR   = 0;
constexpr size_t O_LOSS = (size_t)T * FD;
constexpr size_t O_ZIDX = O_LOSS + 1;
constexpr size_t O_QIDX = O_ZIDX + T;
constexpr size_t O_ZCB  = O_QIDX + T;
constexpr size_t O_QCB  = O_ZCB + (size_t)NZ * ZD;

// ---- workspace layout (floats) ----
constexpr size_t W_ZE   = 0;                               // [T, ZD]
constexpr size_t W_QC   = W_ZE + (size_t)T * ZD;           // [T, QD]
constexpr size_t W_ESZ  = W_QC + (size_t)T * QD;           // [NZ, ZD] segment sums
constexpr size_t W_CSZ  = W_ESZ + (size_t)NZ * ZD;         // [NZ] counts
constexpr size_t W_NCSZ = W_CSZ + NZ;                      // [NZ] new ema counts
constexpr size_t W_NCBZ = W_NCSZ + NZ;                     // [NZ, ZD] new cb (pre-replace)
constexpr size_t W_RNZ  = W_NCBZ + (size_t)NZ * ZD;        // [NZ] row norms
constexpr size_t W_DDZ  = W_RNZ + NZ;                      // [NZ] dead flags
constexpr size_t W_NRZ  = W_DDZ + NZ;                      // [NZ] nearest (int)
constexpr size_t W_CNZ  = W_NRZ + NZ;                      // [NZ] |cb|^2
constexpr size_t W_ESQ  = W_CNZ + NZ;
constexpr size_t W_CSQ  = W_ESQ + (size_t)NQ * QD;
constexpr size_t W_NCSQ = W_CSQ + NQ;
constexpr size_t W_NCBQ = W_NCSQ + NQ;
constexpr size_t W_RNQ  = W_NCBQ + (size_t)NQ * QD;
constexpr size_t W_DDQ  = W_RNQ + NQ;
constexpr size_t W_NRQ  = W_DDQ + NQ;
constexpr size_t W_CNQ  = W_NRQ + NQ;
constexpr size_t W_WN   = W_CNQ + NQ;                      // [QD, ZD] lip weights (f32)
constexpr size_t W_ACC  = W_WN + (size_t)QD * ZD;          // 8 accumulators

// fragment-linearized bf16 operand buffers (size in float slots, 64B aligned)
constexpr size_t bfslots(size_t elems) { return (elems / 2 + 15) & ~(size_t)15; }
constexpr size_t W_SW_E1  = W_ACC + 16;                       // enc W1  512x64
constexpr size_t W_SW_E2  = W_SW_E1  + bfslots(512 * 64);     // enc W2  64x128
constexpr size_t W_SW_WZ  = W_SW_E2  + bfslots(64 * 128);     // Wz      128x128
constexpr size_t W_SW_ZCB = W_SW_WZ  + bfslots(128 * 128);    // z_cb^T  128x1024
constexpr size_t W_SW_WN  = W_SW_ZCB + bfslots(1024 * 128);   // Wn^T    128x64
constexpr size_t W_SW_T1  = W_SW_WN  + bfslots(128 * 64);     // t_W1    64x128
constexpr size_t W_SW_T2  = W_SW_T1  + bfslots(64 * 128);     // t_W2    128x64
constexpr size_t W_SW_QCB = W_SW_T2  + bfslots(128 * 64);     // q_cb^T  64x512
constexpr size_t W_SW_D1  = W_SW_QCB + bfslots(512 * 64);     // dec W1  64x64
constexpr size_t W_SW_D2  = W_SW_D1  + bfslots(64 * 64);      // dec W2  64x128
constexpr size_t W_SW_WO  = W_SW_D2  + bfslots(64 * 128);     // Wo      128x512
constexpr size_t W_SW_NCZ = W_SW_WO  + bfslots(128 * 512);    // new z cb^T
constexpr size_t W_SW_NCQ = W_SW_NCZ + bfslots(1024 * 128);   // new q cb^T

// ======================= fragment helpers (wave32 WMMA) ====================
DEVINL bf16_t to_bf(float f) { return (bf16_t)f; }
DEVINL float geluf(float x) { return 0.5f * x * (1.0f + erff(x * 0.70710678f)); }
DEVINL float sigm(float x) { return 1.0f / (1.0f + expf(-x)); }
DEVINL v8f vzero() { v8f z = {}; return z; }

// 16-bit A layout (ISA 7.12.2): element e -> k within a 32-wide K block
DEVINL int kpat(int hi, int e) {
  int v = e >> 1;
  return ((v & 4) ? 16 : 0) + hi * 8 + ((v & 3) << 1) + (e & 1);
}

// A fragment from row-major f32 (rare path: amortized over many n-tiles)
DEVINL v16bf load_a(const float* src, int ld, int m0, int k0, int lane) {
  const int r = lane & 15, hi = lane >> 4;
  const float* p = src + (size_t)(m0 + r) * ld + k0;
  v16bf a;
#pragma unroll
  for (int e = 0; e < 16; ++e) a[e] = to_bf(p[kpat(hi, e)]);
  return a;
}
DEVINL v16bf load_a_gather(const float* src, int ld, int grow, int k0, int lane) {
  const int hi = lane >> 4;
  const float* p = src + (size_t)grow * ld + k0;
  v16bf a;
#pragma unroll
  for (int e = 0; e < 16; ++e) a[e] = to_bf(p[kpat(hi, e)]);
  return a;
}
// B fragment from fragment-linearized bf16 buffer: two b128 loads, zero VALU
DEVINL v16bf load_bswz(const bf16_t* buf, int KT, int kt, int nt, int lane) {
  return *(const v16bf*)(buf + ((((size_t)nt * KT) + kt) * 32 + lane) * 16);
}
// bf16 A-swizzled LDS staging: store one scalar (m, c) / load one fragment
DEVINL void store_stage(bf16_t* st, int m, int c, float v) {
  const int kt = c >> 5, k = c & 31;
  const int hi2 = (k >> 3) & 1;
  const int e2 = ((k >> 1) & 8) | (k & 7);
  st[(kt * 32 + hi2 * 16 + m) * 16 + e2] = to_bf(v);
}
DEVINL float load_stage_elem(const bf16_t* st, int m, int c) {
  const int kt = c >> 5, k = c & 31;
  const int hi2 = (k >> 3) & 1;
  const int e2 = ((k >> 1) & 8) | (k & 7);
  return (float)st[(kt * 32 + hi2 * 16 + m) * 16 + e2];
}
DEVINL v16bf load_stage(const bf16_t* st, int kt, int lane) {
  return *(const v16bf*)(st + ((size_t)(kt * 32 + lane)) * 16);
}
DEVINL v8f wmma_bf(v16bf a, v16bf b, v8f c) {
  return __builtin_amdgcn_wmma_f32_16x16x32_bf16(false, a, false, b, (short)0, c,
                                                 false, false);
}

// ================================ kernels ==================================
__global__ void k_zero(float* p, int n) {
  int i = blockIdx.x * blockDim.x + threadIdx.x;
  if (i < n) p[i] = 0.f;
}

// pre-swizzle: B[k][n] from row-major W[K][N]
__global__ void k_swz_kn(const float* __restrict__ w, int K, int N,
                         bf16_t* __restrict__ out) {
  const int KT = K >> 5;
  const int total = KT * (N >> 4) * 32;
  const int t = blockIdx.x * blockDim.x + threadIdx.x;
  if (t >= total) return;
  const int lane = t & 31, g = t >> 5;
  const int kt = g % KT, nt = g / KT;
  const int hi = lane >> 4, nn = lane & 15;
  bf16_t* o = out + (size_t)t * 16;
#pragma unroll
  for (int e = 0; e < 16; ++e)
    o[e] = to_bf(w[(size_t)(kt * 32 + kpat(hi, e)) * N + nt * 16 + nn]);
}
// pre-swizzle: B columns = rows of M[N][K] (codebooks, Wn^T)
__global__ void k_swz_nk(const float* __restrict__ m, int nrows, int K,
                         bf16_t* __restrict__ out) {
  const int KT = K >> 5;
  const int total = KT * (nrows >> 4) * 32;
  const int t = blockIdx.x * blockDim.x + threadIdx.x;
  if (t >= total) return;
  const int lane = t & 31, g = t >> 5;
  const int kt = g % KT, nt = g / KT;
  const int hi = lane >> 4, nn = lane & 15;
  bf16_t* o = out + (size_t)t * 16;
#pragma unroll
  for (int e = 0; e < 16; ++e)
    o[e] = to_bf(m[(size_t)(nt * 16 + nn) * K + kt * 32 + kpat(hi, e)]);
}

__global__ void k_rownorm(const float* __restrict__ m, float* __restrict__ out, int D) {
  __shared__ float sh[128];
  const int n = blockIdx.x, d = threadIdx.x;
  float v = m[(size_t)n * D + d];
  sh[d] = v * v;
  __syncthreads();
  for (int s = blockDim.x >> 1; s > 0; s >>= 1) {
    if (d < s) sh[d] += sh[d + s];
    __syncthreads();
  }
  if (d == 0) out[n] = sh[0];
}

// Wn = lip_W * min(1, softplus(ci)/sum|row|)
__global__ void k_wn(const float* __restrict__ lipW, const float* __restrict__ lipci,
                     float* __restrict__ wn) {
  __shared__ float sh[128];
  __shared__ float scl;
  const int n = blockIdx.x, d = threadIdx.x;
  const float w = lipW[n * ZD + d];
  sh[d] = fabsf(w);
  __syncthreads();
  for (int s = 64; s > 0; s >>= 1) {
    if (d < s) sh[d] += sh[d + s];
    __syncthreads();
  }
  if (d == 0) {
    const float ci = lipci[n];
    const float sp = ci > 20.f ? ci : log1pf(expf(ci));
    scl = fminf(1.f, sp / sh[0]);
  }
  __syncthreads();
  wn[n * ZD + d] = w * scl;
}

// encoder chain: x -> gelu -> gelu -> ze ; Wz block staged in LDS via TDM
__global__ __launch_bounds__(128) void k_encoder(
    const float* __restrict__ x, const bf16_t* __restrict__ sw1,
    const float* __restrict__ b1, const bf16_t* __restrict__ sw2,
    const float* __restrict__ b2, const bf16_t* __restrict__ swwz,
    const float* __restrict__ bz, float* __restrict__ ze) {
  __shared__ __attribute__((aligned(32))) bf16_t stg[4][16 * HID];  // 16 KB
  __shared__ __attribute__((aligned(32))) bf16_t wzl[128 * 128];    // 32 KB
  const int lane = threadIdx.x & 31, wave = threadIdx.x >> 5;
  const int hi = lane >> 4, nn = lane & 15;
  const int m0 = (blockIdx.x * 4 + wave) * 16;
  bf16_t* st = stg[wave];

#if USE_TDM
  if (wave == 0) {
    const unsigned long long ga = (unsigned long long)(uintptr_t)swwz;
    const unsigned int lo = (unsigned int)(uintptr_t)(void*)wzl;
    // D# group0: count=1 | lds_addr | global_addr | type=2
    v4u g0 = {1u, lo, (unsigned int)ga,
              (unsigned int)((ga >> 32) & 0x01FFFFFFu) | (2u << 30)};
    // D# group1: data_size=2B; tensor_dim0=tile_dim0=16384; dim1=tile_dim1=1
    v8i g1 = {(int)(1u << 16), (int)0x40000000u, (int)0x00010000u,
              (int)0x40000000u, 1, 16384, 0, 0};
    v4i gz = {0, 0, 0, 0};
#if __clang_major__ >= 23
    v8i gz8 = {0, 0, 0, 0, 0, 0, 0, 0};
    __builtin_amdgcn_tensor_load_to_lds(g0, g1, gz, gz, gz8, 0);
#else
    __builtin_amdgcn_tensor_load_to_lds(g0, g1, gz, gz, 0);
#endif
    __builtin_amdgcn_s_wait_tensorcnt(0);
  }
  __syncthreads();
#else
  for (int i = threadIdx.x; i < (128 * 128) / 8; i += blockDim.x)
    ((uint4*)wzl)[i] = ((const uint4*)swwz)[i];
  __syncthreads();
#endif

  __builtin_prefetch(x + (size_t)(m0 + nn) * FD + hi * 256, 0, 1);

  // h1 = gelu(x @ W1 + b1) : [16,64]
  v8f acc4[4];
#pragma unroll
  for (int i = 0; i < 4; ++i) acc4[i] = vzero();
  for (int kt = 0; kt < FD / 32; ++kt) {
    v16bf a = load_a(x, FD, m0, kt * 32, lane);
#pragma unroll
    for (int nt = 0; nt < 4; ++nt)
      acc4[nt] = wmma_bf(a, load_bswz(sw1, 16, kt, nt, lane), acc4[nt]);
  }
#pragma unroll
  for (int nt = 0; nt < 4; ++nt) {
    const float bb = b1[nt * 16 + nn];
#pragma unroll
    for (int r = 0; r < 8; ++r)
      store_stage(st, r + 8 * hi, nt * 16 + nn, geluf(acc4[nt][r] + bb));
  }
  // h = gelu(h1 @ W2 + b2) : [16,128]
  v16bf a1[2];
#pragma unroll
  for (int kt = 0; kt < 2; ++kt) a1[kt] = load_stage(st, kt, lane);
  v8f h8[8];
#pragma unroll
  for (int nt = 0; nt < 8; ++nt) {
    v8f acc = vzero();
#pragma unroll
    for (int kt = 0; kt < 2; ++kt)
      acc = wmma_bf(a1[kt], load_bswz(sw2, 2, kt, nt, lane), acc);
    const float bb = b2[nt * 16 + nn];
#pragma unroll
    for (int r = 0; r < 8; ++r) acc[r] = geluf(acc[r] + bb);
    h8[nt] = acc;
  }
#pragma unroll
  for (int nt = 0; nt < 8; ++nt)
#pragma unroll
    for (int r = 0; r < 8; ++r)
      store_stage(st, r + 8 * hi, nt * 16 + nn, h8[nt][r]);
  // ze = h @ Wz + bz : [16,128]  (B fragments from LDS-resident Wz)
  v16bf a2[4];
#pragma unroll
  for (int kt = 0; kt < 4; ++kt) a2[kt] = load_stage(st, kt, lane);
#pragma unroll
  for (int nt = 0; nt < 8; ++nt) {
    v8f acc = vzero();
#pragma unroll
    for (int kt = 0; kt < 4; ++kt)
      acc = wmma_bf(a2[kt], load_bswz(wzl, 4, kt, nt, lane), acc);
    const float bb = bz[nt * 16 + nn];
#pragma unroll
    for (int r = 0; r < 8; ++r)
      ze[(size_t)(m0 + r + 8 * hi) * ZD + nt * 16 + nn] = acc[r] + bb;
  }
}

// VQ: nearest code (WMMA dot on swizzled bf16 codebook), atomics, SSE
template <int D, int N>
__global__ __launch_bounds__(256) void k_vq(
    const float* __restrict__ zmat, const float* __restrict__ cb,
    const bf16_t* __restrict__ cbsw, const float* __restrict__ cbnorm,
    float* __restrict__ cs, float* __restrict__ es, float* __restrict__ out_idx,
    float* __restrict__ sse_acc) {
  __shared__ float bsum;
  const int lane = threadIdx.x & 31, wave = threadIdx.x >> 5;
  const int hi = lane >> 4, nn = lane & 15;
  const int m0 = (blockIdx.x * 8 + wave) * 16;
  if (threadIdx.x == 0) bsum = 0.f;
  __syncthreads();
  constexpr int KT = D / 32;
  v16bf a[KT];
#pragma unroll
  for (int kt = 0; kt < KT; ++kt) a[kt] = load_a(zmat, D, m0, kt * 32, lane);
  float bd[8]; int bi[8];
#pragma unroll
  for (int r = 0; r < 8; ++r) { bd[r] = 3.4e38f; bi[r] = 0; }
  for (int nt = 0; nt < N / 16; ++nt) {
    v8f acc = vzero();
#pragma unroll
    for (int kt = 0; kt < KT; ++kt)
      acc = wmma_bf(a[kt], load_bswz(cbsw, KT, kt, nt, lane), acc);
    const float cn = cbnorm[nt * 16 + nn];
#pragma unroll
    for (int r = 0; r < 8; ++r) {
      const float dist = cn - 2.f * acc[r];   // |ze|^2 constant per row: dropped
      if (dist < bd[r]) { bd[r] = dist; bi[r] = nt * 16 + nn; }
    }
  }
  float ssel = 0.f;
#pragma unroll
  for (int r = 0; r < 8; ++r) {
    float d = bd[r]; int i = bi[r];
#pragma unroll
    for (int off = 8; off > 0; off >>= 1) {   // 16-lane butterfly (wave32 halves)
      const float od = __shfl_xor(d, off, 16);
      const int oi = __shfl_xor(i, off, 16);
      if (od < d || (od == d && oi < i)) { d = od; i = oi; }
    }
    const int row = m0 + r + 8 * hi;
    float part = 0.f;
    for (int d0 = nn; d0 < D; d0 += 16) {
      const float zv = zmat[(size_t)row * D + d0];
      const float cv = cb[(size_t)i * D + d0];
      const float df = zv - cv;
      part += df * df;
      atomicAdd(&es[(size_t)i * D + d0], zv);
    }
#pragma unroll
    for (int off = 8; off > 0; off >>= 1) part += __shfl_xor(part, off, 16);
    if (nn == 0) {
      ssel += part;
      atomicAdd(&cs[i], 1.f);
      out_idx[row] = (float)i;
    }
  }
  atomicAdd(&bsum, ssel);
  __syncthreads();
  if (threadIdx.x == 0) atomicAdd(sse_acc, bsum);
}

__global__ void k_newcs(const float* __restrict__ emacs, const float* __restrict__ cs,
                        float* __restrict__ newcs, float* __restrict__ nsum, int N) {
  __shared__ float sh[1024];
  const int n = threadIdx.x;
  float v = 0.f;
  if (n < N) {
    v = DECAY * emacs[n] + (1.f - DECAY) * cs[n];
    newcs[n] = v;
  }
  sh[n] = v;
  __syncthreads();
  for (int s = 512; s > 0; s >>= 1) {
    if (n < s) sh[n] += sh[n + s];
    __syncthreads();
  }
  if (n == 0) *nsum = sh[0];
}

__global__ void k_newcb(const float* __restrict__ emacb, const float* __restrict__ es,
                        const float* __restrict__ newcs, const float* __restrict__ nsum,
                        const float* __restrict__ usage, const float* __restrict__ cs,
                        float thresh, int N, int D, float* __restrict__ newcb,
                        float* __restrict__ rnorm, float* __restrict__ dead) {
  __shared__ float sh[128];
  const int n = blockIdx.x, d = threadIdx.x;
  const float necb =
      DECAY * emacb[(size_t)n * D + d] + (1.f - DECAY) * es[(size_t)n * D + d];
  const float denom = (newcs[n] + EPSV) / (nsum[0] + (float)N * EPSV);
  const float v = necb / denom;
  newcb[(size_t)n * D + d] = v;
  sh[d] = v * v;
  __syncthreads();
  for (int s = blockDim.x >> 1; s > 0; s >>= 1) {
    if (d < s) sh[d] += sh[d + s];
    __syncthreads();
  }
  if (d == 0) {
    rnorm[n] = sh[0];
    dead[n] = ((usage[n] + cs[n]) < thresh) ? 1.f : 0.f;
  }
}

// pairwise nearest-alive search over the updated codebook (WMMA)
template <int N, int D>
__global__ __launch_bounds__(256) void k_nearest(
    const float* __restrict__ ncb, const bf16_t* __restrict__ ncbsw,
    const float* __restrict__ rnorm, const float* __restrict__ dead,
    int* __restrict__ nearest) {
  const int lane = threadIdx.x & 31, wave = threadIdx.x >> 5;
  const int hi = lane >> 4, nn = lane & 15;
  const int m0 = (blockIdx.x * 8 + wave) * 16;
  constexpr int KT = D / 32;
  v16bf a[KT];
#pragma unroll
  for (int kt = 0; kt < KT; ++kt) a[kt] = load_a(ncb, D, m0, kt * 32, lane);
  float bd[8]; int bi[8];
#pragma unroll
  for (int r = 0; r < 8; ++r) { bd[r] = 3.4e38f; bi[r] = 0; }
  for (int nt = 0; nt < N / 16; ++nt) {
    v8f acc = vzero();
#pragma unroll
    for (int kt = 0; kt < KT; ++kt)
      acc = wmma_bf(a[kt], load_bswz(ncbsw, KT, kt, nt, lane), acc);
    const float cn = rnorm[nt * 16 + nn];
    const bool isdead = dead[nt * 16 + nn] > 0.5f;
#pragma unroll
    for (int r = 0; r < 8; ++r) {
      const float dist = isdead ? 3.0e38f : (cn - 2.f * acc[r]);
      if (dist < bd[r]) { bd[r] = dist; bi[r] = nt * 16 + nn; }
    }
  }
#pragma unroll
  for (int r = 0; r < 8; ++r) {
    float d = bd[r]; int i = bi[r];
#pragma unroll
    for (int off = 8; off > 0; off >>= 1) {
      const float od = __shfl_xor(d, off, 16);
      const int oi = __shfl_xor(i, off, 16);
      if (od < d || (od == d && oi < i)) { d = od; i = oi; }
    }
    if (nn == 0) nearest[m0 + r + 8 * hi] = i;
  }
}

__global__ void k_replace(const float* __restrict__ newcb, const float* __restrict__ dead,
                          const int* __restrict__ nearest, int N, int D,
                          float* __restrict__ out) {
  const int t = blockIdx.x * blockDim.x + threadIdx.x;
  if (t >= N * D) return;
  const int n = t / D, d = t % D;
  const int src = dead[n] > 0.5f ? nearest[n] : n;
  out[t] = newcb[(size_t)src * D + d];
}

// lipschitz head (zq gathered from codebook) + time-prediction MLP + time loss
__global__ __launch_bounds__(128) void k_lip_time(
    const float* __restrict__ zcb, const float* __restrict__ zidxf,
    const bf16_t* __restrict__ swwn, const float* __restrict__ lipb,
    const bf16_t* __restrict__ swt1, const float* __restrict__ tb1,
    const bf16_t* __restrict__ swt2, const float* __restrict__ tb2,
    const float* __restrict__ tW3, const float* __restrict__ tb3,
    float* __restrict__ qc, float* __restrict__ tsse_acc) {
  __shared__ __attribute__((aligned(32))) bf16_t stg[4][16 * HID];
  __shared__ float bsum;
  const int lane = threadIdx.x & 31, wave = threadIdx.x >> 5;
  const int hi = lane >> 4, nn = lane & 15;
  const int m0 = (blockIdx.x * 4 + wave) * 16;
  bf16_t* st = stg[wave];
  if (threadIdx.x == 0) bsum = 0.f;
  __syncthreads();

  const int grow = (int)zidxf[m0 + nn];     // zq row = z_cb[z_idx[row]]
  v16bf a[4];
#pragma unroll
  for (int kt = 0; kt < 4; ++kt) a[kt] = load_a_gather(zcb, ZD, grow, kt * 32, lane);
  // q_cont = sigmoid(zq @ Wn^T + lip_b) : [16,64]
#pragma unroll
  for (int nt = 0; nt < 4; ++nt) {
    v8f acc = vzero();
#pragma unroll
    for (int kt = 0; kt < 4; ++kt)
      acc = wmma_bf(a[kt], load_bswz(swwn, 4, kt, nt, lane), acc);
    const float bb = lipb[nt * 16 + nn];
#pragma unroll
    for (int r = 0; r < 8; ++r) {
      const float v = sigm(acc[r] + bb);
      store_stage(st, r + 8 * hi, nt * 16 + nn, v);
      qc[(size_t)(m0 + r + 8 * hi) * QD + nt * 16 + nn] = v;
    }
  }
  // tp1 = sigmoid(qc @ tW1 + tb1) : [16,128]
  v16bf a1[2];
#pragma unroll
  for (int kt = 0; kt < 2; ++kt) a1[kt] = load_stage(st, kt, lane);
  v8f h8[8];
#pragma unroll
  for (int nt = 0; nt < 8; ++nt) {
    v8f acc = vzero();
#pragma unroll
    for (int kt = 0; kt < 2; ++kt)
      acc = wmma_bf(a1[kt], load_bswz(swt1, 2, kt, nt, lane), acc);
    const float bb = tb1[nt * 16 + nn];
#pragma unroll
    for (int r = 0; r < 8; ++r) acc[r] = sigm(acc[r] + bb);
    h8[nt] = acc;
  }
#pragma unroll
  for (int nt = 0; nt < 8; ++nt)
#pragma unroll
    for (int r = 0; r < 8; ++r)
      store_stage(st, r + 8 * hi, nt * 16 + nn, h8[nt][r]);
  // tp2 = sigmoid(tp1 @ tW2 + tb2) : [16,64]
  v16bf a2[4];
#pragma unroll
  for (int kt = 0; kt < 4; ++kt) a2[kt] = load_stage(st, kt, lane);
  v8f t2[4];
#pragma unroll
  for (int nt = 0; nt < 4; ++nt) {
    v8f acc = vzero();
#pragma unroll
    for (int kt = 0; kt < 4; ++kt)
      acc = wmma_bf(a2[kt], load_bswz(swt2, 4, kt, nt, lane), acc);
    const float bb = tb2[nt * 16 + nn];
#pragma unroll
    for (int r = 0; r < 8; ++r) acc[r] = sigm(acc[r] + bb);
    t2[nt] = acc;
  }
#pragma unroll
  for (int nt = 0; nt < 4; ++nt)
#pragma unroll
    for (int r = 0; r < 8; ++r)
      store_stage(st, r + 8 * hi, nt * 16 + nn, t2[nt][r]);
  // tpred = tp2 @ tW3 + tb3; time SSE (2 lanes per row)
  const int rrow = lane >> 1, half = lane & 1;
  float part = 0.f;
  for (int d0 = half * 32; d0 < half * 32 + 32; ++d0)
    part += load_stage_elem(st, rrow, d0) * tW3[d0];
  part += __shfl_xor(part, 1, 32);
  if (half == 0) {
    const int gr = m0 + rrow;
    const float tt = (float)(gr & (Ss - 1)) * (1.f / 1023.f);
    const float df = (part + tb3[0]) - tt;
    atomicAdd(&bsum, df * df);
  }
  __syncthreads();
  if (threadIdx.x == 0) atomicAdd(tsse_acc, bsum);
}

// decoder: qq (=q_cb[q_idx], straight-through) -> gelu -> gelu -> x_recon + SSE
__global__ __launch_bounds__(128) void k_decoder(
    const float* __restrict__ qcb, const float* __restrict__ qidxf,
    const bf16_t* __restrict__ swd1, const float* __restrict__ db1,
    const bf16_t* __restrict__ swd2, const float* __restrict__ db2,
    const bf16_t* __restrict__ swwo, const float* __restrict__ bo,
    const float* __restrict__ x, float* __restrict__ xr,
    float* __restrict__ recon_acc) {
  __shared__ __attribute__((aligned(32))) bf16_t stg[4][16 * HID];
  __shared__ float bsum;
  const int lane = threadIdx.x & 31, wave = threadIdx.x >> 5;
  const int hi = lane >> 4, nn = lane & 15;
  const int m0 = (blockIdx.x * 4 + wave) * 16;
  bf16_t* st = stg[wave];
  if (threadIdx.x == 0) bsum = 0.f;
  __syncthreads();

  const int grow = (int)qidxf[m0 + nn];
  v16bf aq[2];
#pragma unroll
  for (int kt = 0; kt < 2; ++kt) aq[kt] = load_a_gather(qcb, QD, grow, kt * 32, lane);
  // dh1 = gelu(qq @ dW1 + db1) : [16,64]
#pragma unroll
  for (int nt = 0; nt < 4; ++nt) {
    v8f acc = vzero();
#pragma unroll
    for (int kt = 0; kt < 2; ++kt)
      acc = wmma_bf(aq[kt], load_bswz(swd1, 2, kt, nt, lane), acc);
    const float bb = db1[nt * 16 + nn];
#pragma unroll
    for (int r = 0; r < 8; ++r)
      store_stage(st, r + 8 * hi, nt * 16 + nn, geluf(acc[r] + bb));
  }
  // dh = gelu(dh1 @ dW2 + db2) : [16,128]
  v16bf a1[2];
#pragma unroll
  for (int kt = 0; kt < 2; ++kt) a1[kt] = load_stage(st, kt, lane);
  v8f h8[8];
#pragma unroll
  for (int nt = 0; nt < 8; ++nt) {
    v8f acc = vzero();
#pragma unroll
    for (int kt = 0; kt < 2; ++kt)
      acc = wmma_bf(a1[kt], load_bswz(swd2, 2, kt, nt, lane), acc);
    const float bb = db2[nt * 16 + nn];
#pragma unroll
    for (int r = 0; r < 8; ++r) acc[r] = geluf(acc[r] + bb);
    h8[nt] = acc;
  }
#pragma unroll
  for (int nt = 0; nt < 8; ++nt)
#pragma unroll
    for (int r = 0; r < 8; ++r)
      store_stage(st, r + 8 * hi, nt * 16 + nn, h8[nt][r]);
  // x_recon = dh @ Wo + bo : [16,512], fused recon SSE
  v16bf a2[4];
#pragma unroll
  for (int kt = 0; kt < 4; ++kt) a2[kt] = load_stage(st, kt, lane);
  float ssel = 0.f;
  for (int nt = 0; nt < 32; ++nt) {
    v8f acc = vzero();
#pragma unroll
    for (int kt = 0; kt < 4; ++kt)
      acc = wmma_bf(a2[kt], load_bswz(swwo, 4, kt, nt, lane), acc);
    const float bb = bo[nt * 16 + nn];
#pragma unroll
    for (int r = 0; r < 8; ++r) {
      const size_t idx = (size_t)(m0 + r + 8 * hi) * FD + nt * 16 + nn;
      const float v = acc[r] + bb;
      xr[idx] = v;
      const float df = v - x[idx];
      ssel += df * df;
    }
  }
  atomicAdd(&bsum, ssel);
  __syncthreads();
  if (threadIdx.x == 0) atomicAdd(recon_acc, bsum);
}

__global__ void k_finalize(const float* __restrict__ acc, float* __restrict__ loss) {
  const float zsse = acc[0], qsse = acc[1], tsse = acc[2], rsse = acc[3];
  const float recon = rsse / ((float)T * (float)FD);
  const float cz = zsse / ((float)T * (float)ZD);   // commit_z == codebook_z numerically
  const float cq = qsse / ((float)T * (float)QD);
  const float tl = tsse / (float)T;
  *loss = recon + 0.25f * (2.f * cz + cq) + tl;
}

// ================================ driver ===================================
extern "C" void kernel_launch(void* const* d_in, const int* in_sizes, int n_in,
                              void* d_out, int out_size, void* d_ws, size_t ws_size,
                              hipStream_t stream) {
  const float* x      = (const float*)d_in[0];
  const float* encW1  = (const float*)d_in[1];
  const float* encb1  = (const float*)d_in[2];
  const float* encW2  = (const float*)d_in[3];
  const float* encb2  = (const float*)d_in[4];
  const float* Wz     = (const float*)d_in[5];
  const float* bz     = (const float*)d_in[6];
  const float* zcb    = (const float*)d_in[7];
  const float* zemacs = (const float*)d_in[8];
  const float* zemacb = (const float*)d_in[9];
  const float* zusage = (const float*)d_in[10];
  const float* lipW   = (const float*)d_in[11];
  const float* lipb   = (const float*)d_in[12];
  const float* lipci  = (const float*)d_in[13];
  const float* tW1    = (const float*)d_in[14];
  const float* tb1    = (const float*)d_in[15];
  const float* tW2    = (const float*)d_in[16];
  const float* tb2    = (const float*)d_in[17];
  const float* tW3    = (const float*)d_in[18];
  const float* tb3    = (const float*)d_in[19];
  const float* qcb    = (const float*)d_in[20];
  const float* qemacs = (const float*)d_in[21];
  const float* qemacb = (const float*)d_in[22];
  const float* qusage = (const float*)d_in[23];
  const float* dW1    = (const float*)d_in[24];
  const float* db1    = (const float*)d_in[25];
  const float* dW2    = (const float*)d_in[26];
  const float* db2    = (const float*)d_in[27];
  const float* Wo     = (const float*)d_in[28];
  const float* bo     = (const float*)d_in[29];

  float* out = (float*)d_out;
  float* ws = (float*)d_ws;
  auto bfp = [&](size_t off) { return (bf16_t*)(ws + off); };
  auto swg = [](int elems) { return (elems / 16 + 255) / 256; };  // threads = elems/16

  k_zero<<<(NZ * ZD + NZ + 255) / 256, 256, 0, stream>>>(ws + W_ESZ, NZ * ZD + NZ);
  k_zero<<<(NQ * QD + NQ + 255) / 256, 256, 0, stream>>>(ws + W_ESQ, NQ * QD + NQ);
  k_zero<<<1, 8, 0, stream>>>(ws + W_ACC, 8);

  k_rownorm<<<NZ, ZD, 0, stream>>>(zcb, ws + W_CNZ, ZD);
  k_rownorm<<<NQ, QD, 0, stream>>>(qcb, ws + W_CNQ, QD);
  k_wn<<<QD, ZD, 0, stream>>>(lipW, lipci, ws + W_WN);

  // one-time bf16 fragment-linearization of all B operands
  k_swz_kn<<<swg(512 * 64), 256, 0, stream>>>(encW1, 512, 64, bfp(W_SW_E1));
  k_swz_kn<<<swg(64 * 128), 256, 0, stream>>>(encW2, 64, 128, bfp(W_SW_E2));
  k_swz_kn<<<swg(128 * 128), 256, 0, stream>>>(Wz, 128, 128, bfp(W_SW_WZ));
  k_swz_nk<<<swg(1024 * 128), 256, 0, stream>>>(zcb, 1024, 128, bfp(W_SW_ZCB));
  k_swz_nk<<<swg(64 * 128), 256, 0, stream>>>(ws + W_WN, 64, 128, bfp(W_SW_WN));
  k_swz_kn<<<swg(64 * 128), 256, 0, stream>>>(tW1, 64, 128, bfp(W_SW_T1));
  k_swz_kn<<<swg(128 * 64), 256, 0, stream>>>(tW2, 128, 64, bfp(W_SW_T2));
  k_swz_nk<<<swg(512 * 64), 256, 0, stream>>>(qcb, 512, 64, bfp(W_SW_QCB));
  k_swz_kn<<<swg(64 * 64), 256, 0, stream>>>(dW1, 64, 64, bfp(W_SW_D1));
  k_swz_kn<<<swg(64 * 128), 256, 0, stream>>>(dW2, 64, 128, bfp(W_SW_D2));
  k_swz_kn<<<swg(128 * 512), 256, 0, stream>>>(Wo, 128, 512, bfp(W_SW_WO));

  k_encoder<<<T / 64, 128, 0, stream>>>(x, bfp(W_SW_E1), encb1, bfp(W_SW_E2), encb2,
                                        bfp(W_SW_WZ), bz, ws + W_ZE);

  k_vq<ZD, NZ><<<T / 128, 256, 0, stream>>>(ws + W_ZE, zcb, bfp(W_SW_ZCB), ws + W_CNZ,
                                            ws + W_CSZ, ws + W_ESZ, out + O_ZIDX,
                                            ws + W_ACC + 0);
  k_newcs<<<1, 1024, 0, stream>>>(zemacs, ws + W_CSZ, ws + W_NCSZ, ws + W_ACC + 4, NZ);
  k_newcb<<<NZ, ZD, 0, stream>>>(zemacb, ws + W_ESZ, ws + W_NCSZ, ws + W_ACC + 4, zusage,
                                 ws + W_CSZ, ZTH, NZ, ZD, ws + W_NCBZ, ws + W_RNZ,
                                 ws + W_DDZ);
  k_swz_nk<<<swg(1024 * 128), 256, 0, stream>>>(ws + W_NCBZ, 1024, 128, bfp(W_SW_NCZ));
  k_nearest<NZ, ZD><<<NZ / 128, 256, 0, stream>>>(ws + W_NCBZ, bfp(W_SW_NCZ), ws + W_RNZ,
                                                  ws + W_DDZ, (int*)(ws + W_NRZ));
  k_replace<<<(NZ * ZD + 255) / 256, 256, 0, stream>>>(ws + W_NCBZ, ws + W_DDZ,
                                                       (int*)(ws + W_NRZ), NZ, ZD,
                                                       out + O_ZCB);

  k_lip_time<<<T / 64, 128, 0, stream>>>(zcb, out + O_ZIDX, bfp(W_SW_WN), lipb,
                                         bfp(W_SW_T1), tb1, bfp(W_SW_T2), tb2, tW3, tb3,
                                         ws + W_QC, ws + W_ACC + 2);

  k_vq<QD, NQ><<<T / 128, 256, 0, stream>>>(ws + W_QC, qcb, bfp(W_SW_QCB), ws + W_CNQ,
                                            ws + W_CSQ, ws + W_ESQ, out + O_QIDX,
                                            ws + W_ACC + 1);
  k_newcs<<<1, 1024, 0, stream>>>(qemacs, ws + W_CSQ, ws + W_NCSQ, ws + W_ACC + 5, NQ);
  k_newcb<<<NQ, QD, 0, stream>>>(qemacb, ws + W_ESQ, ws + W_NCSQ, ws + W_ACC + 5, qusage,
                                 ws + W_CSQ, QTH, NQ, QD, ws + W_NCBQ, ws + W_RNQ,
                                 ws + W_DDQ);
  k_swz_nk<<<swg(512 * 64), 256, 0, stream>>>(ws + W_NCBQ, 512, 64, bfp(W_SW_NCQ));
  k_nearest<NQ, QD><<<NQ / 128, 256, 0, stream>>>(ws + W_NCBQ, bfp(W_SW_NCQ), ws + W_RNQ,
                                                  ws + W_DDQ, (int*)(ws + W_NRQ));
  k_replace<<<(NQ * QD + 255) / 256, 256, 0, stream>>>(ws + W_NCBQ, ws + W_DDQ,
                                                       (int*)(ws + W_NRQ), NQ, QD,
                                                       out + O_QCB);

  k_decoder<<<T / 64, 128, 0, stream>>>(qcb, out + O_QIDX, bfp(W_SW_D1), db1,
                                        bfp(W_SW_D2), db2, bfp(W_SW_WO), bo, x,
                                        out + O_XR, ws + W_ACC + 3);
  k_finalize<<<1, 1, 0, stream>>>(ws + W_ACC, out + O_LOSS);

  (void)in_sizes; (void)n_in; (void)out_size; (void)ws_size;
}
